// DCNFusionHead_26259430048542
// MI455X (gfx1250) — matrix operand order repaired and checked by reading, not compiled, not run on hardware
//
#include <hip/hip_runtime.h>
#include <cstdint>
#include <cstddef>

typedef __attribute__((ext_vector_type(16))) __bf16 v16bf;
typedef __attribute__((ext_vector_type(8)))  float  v8f;

#define BATCH 2
#define HH 256
#define WW 256
#define HW 65536
#define NPIX 131072   // BATCH*HH*WW

// ---------------------------------------------------------------------------
// Layout conversion: NCHW f32 -> NHWC bf16 (optionally into a wider buffer)
// ---------------------------------------------------------------------------
__global__ void nchw_to_nhwc_bf16_k(const float* __restrict__ src, __bf16* __restrict__ dst,
                                    int C, int dstStride, int dstOff, long total) {
  long i = (long)blockIdx.x * blockDim.x + threadIdx.x;
  if (i >= total) return;
  int  c   = (int)(i % C);
  long pix = i / C;                 // b*HW + hw
  long b   = pix >> 16;
  long hw  = pix & 65535;
  dst[pix * dstStride + dstOff + c] = (__bf16)src[(b * C + c) * (long)HW + hw];
}

// OIHW f32 -> bf16 [Np][K], K ordered (tap, cin), zero pad n >= O
__global__ void conv_weights_bf16_k(const float* __restrict__ src, __bf16* __restrict__ dst,
                                    int O, int I, int taps, int Np) {
  int  K = I * taps;
  long total = (long)Np * K;
  long i = (long)blockIdx.x * blockDim.x + threadIdx.x;
  if (i >= total) return;
  int n = (int)(i / K), k = (int)(i % K);
  int tap = k / I, c = k - tap * I;
  float v = (n < O) ? src[((long)n * I + c) * taps + tap] : 0.f;
  dst[i] = (__bf16)v;
}

// ---------------------------------------------------------------------------
// CDNA5 async helpers (inline asm; ASYNCcnt-tracked LDS fill)
// ---------------------------------------------------------------------------
__device__ __forceinline__ void async_lds_b128(unsigned ldsAddr, const void* gaddr) {
  asm volatile("global_load_async_to_lds_b128 %0, %1, off"
               :: "v"(ldsAddr), "v"(gaddr) : "memory");
}
__device__ __forceinline__ void wait_asynccnt0() {
  asm volatile("s_wait_asynccnt 0x0" ::: "memory");
}

// ---------------------------------------------------------------------------
// Generic implicit-GEMM conv (3x3 pad1 when taps==9, 1x1 pad0 when taps==1)
// NHWC bf16 in, bf16 WMMA, f32 accumulate, BN/bias/ReLU folded epilogue.
// Block: 256 threads = 8 waves; 128 pixels x Np channels per block.
// Double-buffered LDS; weights staged via global_load_async_to_lds_b128.
// ---------------------------------------------------------------------------
template<int NT>
__global__ __launch_bounds__(256)
void conv_wmma_k(const __bf16* __restrict__ in, int Cin, int taps,
                 const __bf16* __restrict__ wgt, int Cout,
                 const float* __restrict__ gw, const float* __restrict__ bw,
                 const float* __restrict__ mw, const float* __restrict__ vw,
                 const float* __restrict__ bias, int relu,
                 __bf16* __restrict__ obf, int sOb,
                 float* __restrict__ of32h, int sOf,
                 float* __restrict__ of32c) {
  __shared__ __align__(16) __bf16 ldsA[2][128 * 32];
  __shared__ __align__(16) __bf16 ldsB[2][64 * 32];
  const int t    = threadIdx.x;
  const int lane = t & 31;
  const int wave = t >> 5;
  const int K       = Cin * taps;
  const int nChunks = K >> 5;
  const int pb = blockIdx.x * 128;

  // A-staging role: 128 rows x 2 segments of 16 bf16
  const int r   = t >> 1, seg = t & 1;
  const int pix = pb + r;
  const int bb  = pix >> 16;
  const int yy  = (pix >> 8) & 255;
  const int xx  = pix & 255;
  // B-staging role
  const int bn = t >> 2, bpart = t & 3;

  v8f acc[NT];
  #pragma unroll
  for (int i = 0; i < NT; ++i) {
    #pragma unroll
    for (int j = 0; j < 8; ++j) acc[i][j] = 0.f;
  }

  auto loadA = [&](int kc, uint4& u0, uint4& u1) {
    const int kbase = kc << 5;
    const int tap   = kbase / Cin;
    const int c0    = kbase - tap * Cin;
    const int ky = (taps == 9) ? (tap / 3 - 1) : 0;
    const int kx = (taps == 9) ? (tap % 3 - 1) : 0;
    const int sy = yy + ky, sx = xx + kx;
    u0 = make_uint4(0u, 0u, 0u, 0u);
    u1 = make_uint4(0u, 0u, 0u, 0u);
    if (sy >= 0 && sy < HH && sx >= 0 && sx < WW) {
      const uint4* gp = (const uint4*)(in + (((long)bb * HW + sy * WW + sx) * Cin + c0 + seg * 16));
      u0 = gp[0]; u1 = gp[1];
    }
  };
  auto storeA = [&](int buf, const uint4& u0, const uint4& u1) {
    uint4* lp = (uint4*)&ldsA[buf][r * 32 + seg * 16];
    lp[0] = u0; lp[1] = u1;
  };
  auto asyncB = [&](int kc, int buf) {
    if (bn < NT * 16) {
      unsigned la = (unsigned)(size_t)&ldsB[buf][bn * 32 + bpart * 8];
      async_lds_b128(la, wgt + (long)bn * K + (kc << 5) + bpart * 8);
    }
  };

  // prologue: stage chunk 0 into buffer 0
  uint4 pa0, pa1;
  loadA(0, pa0, pa1);
  asyncB(0, 0);
  storeA(0, pa0, pa1);
  wait_asynccnt0();
  __syncthreads();

  for (int kc = 0; kc < nChunks; ++kc) {
    const int  cur  = kc & 1, nxt = cur ^ 1;
    const bool more = (kc + 1) < nChunks;
    if (more) { loadA(kc + 1, pa0, pa1); asyncB(kc + 1, nxt); }
    // ---- WMMA on current buffers ----
    v16bf afrag;
    {
      const __bf16* ap = &ldsA[cur][(wave * 16 + (lane & 15)) * 32];
      const int klo = (lane >> 4) * 8;
      #pragma unroll
      for (int j = 0; j < 8; ++j) { afrag[j] = ap[klo + j]; afrag[8 + j] = ap[16 + klo + j]; }
    }
    #pragma unroll
    for (int nt = 0; nt < NT; ++nt) {
      v16bf bfrag;
      const __bf16* bp = &ldsB[cur][(nt * 16 + (lane & 15)) * 32 + (lane >> 4) * 8];
      #pragma unroll
      for (int j = 0; j < 8; ++j) { bfrag[j] = bp[j]; bfrag[8 + j] = bp[16 + j]; }
      acc[nt] = __builtin_amdgcn_wmma_f32_16x16x32_bf16(false, afrag, false, bfrag,
                                                        (short)0, acc[nt], false, false);
    }
    if (more) storeA(nxt, pa0, pa1);
    wait_asynccnt0();
    __syncthreads();
  }

  // ---- epilogue: scale/shift (BN fold) + ReLU + stores ----
  #pragma unroll
  for (int nt = 0; nt < NT; ++nt) {
    const int n = nt * 16 + (lane & 15);
    if (n >= Cout) continue;
    float sc = 1.f, sh = 0.f;
    if (bias) sh = bias[n];
    if (gw) { float inv = gw[n] * rsqrtf(vw[n] + 1e-5f); sh = (sh - mw[n]) * inv + bw[n]; sc = inv; }
    #pragma unroll
    for (int rr = 0; rr < 8; ++rr) {
      const int m = (lane >> 4) * 8 + rr;
      const int p = pb + wave * 16 + m;
      float val = acc[nt][rr] * sc + sh;
      if (relu) val = fmaxf(val, 0.f);
      if (obf)   obf[(long)p * sOb + n] = (__bf16)val;
      if (of32h) of32h[(long)p * sOf + n] = val;
      if (of32c) {
        const int pbB = p >> 16, hw = p & 65535;
        of32c[((long)pbB * Cout + n) * HW + hw] = val;
      }
    }
  }
}

// ---------------------------------------------------------------------------
// Deformable conv 3x3: bilinear gather into LDS A, then WMMA (Cin=Cout=64)
// Double-buffered; weights via async-to-LDS.
// ---------------------------------------------------------------------------
__global__ __launch_bounds__(256)
void dcn_wmma_k(const __bf16* __restrict__ feat, const float* __restrict__ off,
                const __bf16* __restrict__ wgt, __bf16* __restrict__ out) {
  __shared__ __align__(16) __bf16 ldsA[2][128 * 32];
  __shared__ __align__(16) __bf16 ldsB[2][64 * 32];
  const int t = threadIdx.x, lane = t & 31, wave = t >> 5;
  const int pb = blockIdx.x * 128;
  const int r = t >> 1, seg = t & 1;
  const int pix = pb + r;
  const int bb = pix >> 16, yy = (pix >> 8) & 255, xx = pix & 255;
  const int bn = t >> 2, bpart = t & 3;

  v8f acc[4];
  #pragma unroll
  for (int i = 0; i < 4; ++i) {
    #pragma unroll
    for (int j = 0; j < 8; ++j) acc[i][j] = 0.f;
  }

  auto gatherA = [&](int kc, float* s) {
    const int tap = kc >> 1, half = kc & 1;
    const int c0 = half * 32 + seg * 16;
    const float dy = off[(long)pix * 32 + tap * 2 + 0];
    const float dx = off[(long)pix * 32 + tap * 2 + 1];
    const float py = (float)yy + (float)(tap / 3 - 1) + dy;
    const float px = (float)xx + (float)(tap % 3 - 1) + dx;
    const float fy = floorf(py), fx = floorf(px);
    const int y0 = (int)fy, x0 = (int)fx;
    const float wy = py - fy, wx = px - fx;
    #pragma unroll
    for (int j = 0; j < 16; ++j) s[j] = 0.f;
    #pragma unroll
    for (int cy = 0; cy < 2; ++cy) {
      #pragma unroll
      for (int cx = 0; cx < 2; ++cx) {
        const int yi = y0 + cy, xi = x0 + cx;
        const float w2 = (cy ? wy : 1.f - wy) * (cx ? wx : 1.f - wx);
        if (yi >= 0 && yi < HH && xi >= 0 && xi < WW) {
          const __bf16* fp = feat + (((long)bb * HW + yi * WW + xi) * 64 + c0);
          #pragma unroll
          for (int j = 0; j < 16; ++j) s[j] += w2 * (float)fp[j];
        }
      }
    }
  };
  auto storeA = [&](int buf, const float* s) {
    #pragma unroll
    for (int j = 0; j < 16; ++j) ldsA[buf][r * 32 + seg * 16 + j] = (__bf16)s[j];
  };
  auto asyncB = [&](int kc, int buf) {
    unsigned la = (unsigned)(size_t)&ldsB[buf][bn * 32 + bpart * 8];
    async_lds_b128(la, wgt + (long)bn * 576 + (kc << 5) + bpart * 8);
  };

  float s[16];
  gatherA(0, s);
  asyncB(0, 0);
  storeA(0, s);
  wait_asynccnt0();
  __syncthreads();

  for (int kc = 0; kc < 18; ++kc) {     // K = 9*64, chunks of 32
    const int  cur = kc & 1, nxt = cur ^ 1;
    const bool more = (kc + 1) < 18;
    if (more) { gatherA(kc + 1, s); asyncB(kc + 1, nxt); }
    v16bf afrag;
    {
      const __bf16* ap = &ldsA[cur][(wave * 16 + (lane & 15)) * 32];
      const int klo = (lane >> 4) * 8;
      #pragma unroll
      for (int j = 0; j < 8; ++j) { afrag[j] = ap[klo + j]; afrag[8 + j] = ap[16 + klo + j]; }
    }
    #pragma unroll
    for (int nt = 0; nt < 4; ++nt) {
      v16bf bfrag;
      const __bf16* bp = &ldsB[cur][(nt * 16 + (lane & 15)) * 32 + (lane >> 4) * 8];
      #pragma unroll
      for (int j = 0; j < 8; ++j) { bfrag[j] = bp[j]; bfrag[8 + j] = bp[16 + j]; }
      acc[nt] = __builtin_amdgcn_wmma_f32_16x16x32_bf16(false, afrag, false, bfrag,
                                                        (short)0, acc[nt], false, false);
    }
    if (more) storeA(nxt, s);
    wait_asynccnt0();
    __syncthreads();
  }
  // epilogue: ReLU, bf16 NHWC stride 64
  #pragma unroll
  for (int nt = 0; nt < 4; ++nt) {
    const int n = nt * 16 + (lane & 15);
    #pragma unroll
    for (int rr = 0; rr < 8; ++rr) {
      const int m = (lane >> 4) * 8 + rr;
      const int p = pb + wave * 16 + m;
      out[(long)p * 64 + n] = (__bf16)fmaxf(acc[nt][rr], 0.f);
    }
  }
}

// ---------------------------------------------------------------------------
// a = sum_c q[c]*k[c]  (16 channels)
// ---------------------------------------------------------------------------
__global__ void qk_dot_k(const __bf16* __restrict__ q, const __bf16* __restrict__ k,
                         float* __restrict__ a) {
  int pix = blockIdx.x * blockDim.x + threadIdx.x;
  if (pix >= NPIX) return;
  float s = 0.f;
  #pragma unroll
  for (int c = 0; c < 16; ++c)
    s += (float)q[(long)pix * 16 + c] * (float)k[(long)pix * 16 + c];
  a[pix] = s;
}

// softmax over 3 branches + weighted sum of v -> fused[..., 256:320]
__global__ void softmax_fuse_k(const float* __restrict__ a0, const float* __restrict__ a1,
                               const float* __restrict__ a2,
                               const __bf16* __restrict__ v0, const __bf16* __restrict__ v1,
                               const __bf16* __restrict__ v2,
                               __bf16* __restrict__ fused) {
  long i = (long)blockIdx.x * blockDim.x + threadIdx.x;
  if (i >= (long)NPIX * 64) return;
  long pix = i >> 6; int c = (int)(i & 63);
  float x0 = a0[pix], x1 = a1[pix], x2 = a2[pix];
  float mx = fmaxf(x0, fmaxf(x1, x2));
  float e0 = __expf(x0 - mx), e1 = __expf(x1 - mx), e2 = __expf(x2 - mx);
  float inv = 1.f / (e0 + e1 + e2);
  float val = (e0 * (float)v0[pix * 64 + c] + e1 * (float)v1[pix * 64 + c]
             + e2 * (float)v2[pix * 64 + c]) * inv;
  fused[pix * 320 + 256 + c] = (__bf16)val;
}

// ---------------------------------------------------------------------------
// Host-side launch helpers
// ---------------------------------------------------------------------------
static void launch_conv(hipStream_t s, const __bf16* in, int Cin, int taps,
                        const __bf16* wgt, int Np, int Cout,
                        const float* g, const float* b, const float* m, const float* v,
                        const float* bias, int relu,
                        __bf16* obf, int sOb, float* ofh, int sOf, float* ofc) {
  dim3 grid(NPIX / 128), blk(256);
  if (Np == 16)
    conv_wmma_k<1><<<grid, blk, 0, s>>>(in, Cin, taps, wgt, Cout, g, b, m, v, bias, relu, obf, sOb, ofh, sOf, ofc);
  else if (Np == 32)
    conv_wmma_k<2><<<grid, blk, 0, s>>>(in, Cin, taps, wgt, Cout, g, b, m, v, bias, relu, obf, sOb, ofh, sOf, ofc);
  else
    conv_wmma_k<4><<<grid, blk, 0, s>>>(in, Cin, taps, wgt, Cout, g, b, m, v, bias, relu, obf, sOb, ofh, sOf, ofc);
}

static void launch_wconv(hipStream_t s, const float* src, __bf16* dst, int O, int I, int taps, int Np) {
  long total = (long)Np * I * taps;
  conv_weights_bf16_k<<<(unsigned)((total + 255) / 256), 256, 0, s>>>(src, dst, O, I, taps, Np);
}

extern "C" void kernel_launch(void* const* d_in, const int* in_sizes, int n_in,
                              void* d_out, int out_size, void* d_ws, size_t ws_size,
                              hipStream_t stream) {
  (void)in_sizes; (void)n_in; (void)out_size; (void)ws_size;
  const float* X = (const float*)d_in[0];
  auto P = [&](int i) { return (const float*)d_in[i]; };

  // ---- workspace bump allocator ----
  uint8_t* wsp = (uint8_t*)d_ws;
  auto alloc = [&](size_t bytes) -> void* {
    void* p = (void*)wsp; wsp += (bytes + 255) & ~(size_t)255; return p;
  };
  __bf16* xb    = (__bf16*)alloc((size_t)NPIX * 256 * 2);
  __bf16* fused = (__bf16*)alloc((size_t)NPIX * 320 * 2);
  __bf16* cf    = (__bf16*)alloc((size_t)NPIX * 64 * 2);
  __bf16* cof   = (__bf16*)alloc((size_t)NPIX * 64 * 2);
  __bf16* ff    = (__bf16*)alloc((size_t)NPIX * 64 * 2);
  __bf16* om    = (__bf16*)alloc((size_t)NPIX * 64 * 2);
  float*  offb  = (float*)alloc((size_t)NPIX * 32 * 4);
  __bf16* fgf   = (__bf16*)alloc((size_t)NPIX * 64 * 2);
  __bf16* ctrf  = (__bf16*)alloc((size_t)NPIX * 64 * 2);
  __bf16* cnrf  = (__bf16*)alloc((size_t)NPIX * 64 * 2);
  __bf16* qb    = (__bf16*)alloc((size_t)NPIX * 16 * 2);
  __bf16* kbuf  = (__bf16*)alloc((size_t)NPIX * 16 * 2);
  float*  a_c   = (float*)alloc((size_t)NPIX * 4);
  float*  a_cn  = (float*)alloc((size_t)NPIX * 4);
  float*  a_f   = (float*)alloc((size_t)NPIX * 4);
  __bf16* v_c   = (__bf16*)alloc((size_t)NPIX * 64 * 2);
  __bf16* v_cn  = (__bf16*)alloc((size_t)NPIX * 64 * 2);
  __bf16* v_f   = (__bf16*)alloc((size_t)NPIX * 64 * 2);
  __bf16* tbuf  = (__bf16*)alloc((size_t)NPIX * 64 * 2);
  // bf16 weight buffers
  __bf16* w_cc  = (__bf16*)alloc((size_t)64 * 2304 * 2);   // center_conv
  __bf16* w_cnc = (__bf16*)alloc((size_t)64 * 2304 * 2);   // corner_conv
  __bf16* w_fgc = (__bf16*)alloc((size_t)64 * 2304 * 2);   // fg_conv
  __bf16* w_chd = (__bf16*)alloc((size_t)16 * 576 * 2);    // center_head
  __bf16* w_cnh = (__bf16*)alloc((size_t)16 * 576 * 2);    // corner_head
  __bf16* w_fgh = (__bf16*)alloc((size_t)16 * 576 * 2);    // fg_head
  __bf16* w_op  = (__bf16*)alloc((size_t)64 * 256 * 2);    // offset_prep
  __bf16* w_off[3]; for (int i = 0; i < 3; ++i) w_off[i] = (__bf16*)alloc((size_t)32 * 64 * 2);
  __bf16* w_dcn[3]; for (int i = 0; i < 3; ++i) w_dcn[i] = (__bf16*)alloc((size_t)64 * 576 * 2);
  __bf16* w_q   = (__bf16*)alloc((size_t)16 * 256 * 2);
  __bf16* w_k[3]; for (int i = 0; i < 3; ++i) w_k[i] = (__bf16*)alloc((size_t)16 * 64 * 2);
  __bf16* w_v[3]; for (int i = 0; i < 3; ++i) w_v[i] = (__bf16*)alloc((size_t)64 * 64 * 2);
  __bf16* w_fh0 = (__bf16*)alloc((size_t)64 * 2880 * 2);
  __bf16* w_fh1 = (__bf16*)alloc((size_t)16 * 576 * 2);
  __bf16* w_hc[5]; for (int i = 0; i < 5; ++i) w_hc[i] = (__bf16*)alloc((size_t)64 * 2880 * 2);
  __bf16* w_hf[5]; for (int i = 0; i < 5; ++i) w_hf[i] = (__bf16*)alloc((size_t)16 * 64 * 2);

  float* out = (float*)d_out;
  const long OFF_REG = 0, OFF_HEIGHT = 262144, OFF_DIM = 393216, OFF_ROT = 786432,
             OFF_VEL = 1048576, OFF_FUSION = 1310720, OFF_CENTER = 1703936,
             OFF_CORNER = 2097152, OFF_FG = 3670016;

  // ---- layout conversion of x (standalone + into fused[...,0:256]) ----
  {
    long tot = (long)NPIX * 256;
    nchw_to_nhwc_bf16_k<<<(unsigned)((tot + 255) / 256), 256, 0, stream>>>(X, xb, 256, 256, 0, tot);
    nchw_to_nhwc_bf16_k<<<(unsigned)((tot + 255) / 256), 256, 0, stream>>>(X, fused, 256, 320, 0, tot);
  }
  // ---- weight conversions ----
  launch_wconv(stream, P(1),  w_cc,  64, 256, 9, 64);
  launch_wconv(stream, P(7),  w_cnc, 64, 256, 9, 64);
  launch_wconv(stream, P(13), w_fgc, 64, 256, 9, 64);
  launch_wconv(stream, P(19), w_chd,  3, 64, 9, 16);
  launch_wconv(stream, P(21), w_cnh, 12, 64, 9, 16);
  launch_wconv(stream, P(23), w_fgh,  3, 64, 9, 16);
  launch_wconv(stream, P(25), w_op,  64, 256, 1, 64);
  const int offW[3] = {31, 34, 37};   // fg_fuse, ctr_fuse, cnr_fuse
  for (int i = 0; i < 3; ++i) {
    launch_wconv(stream, P(offW[i]),     w_off[i], 18, 64, 1, 32);
    launch_wconv(stream, P(offW[i] + 2), w_dcn[i], 64, 64, 9, 64);
  }
  launch_wconv(stream, P(40), w_q, 16, 256, 1, 16);
  const int kvW[3] = {46, 58, 70};    // center, corner, fg
  for (int i = 0; i < 3; ++i) {
    launch_wconv(stream, P(kvW[i]),     w_k[i], 16, 64, 1, 16);
    launch_wconv(stream, P(kvW[i] + 6), w_v[i], 64, 64, 1, 64);
  }
  launch_wconv(stream, P(82), w_fh0, 64, 320, 9, 64);
  launch_wconv(stream, P(88), w_fh1,  3, 64, 9, 16);
  const int hfOc[5] = {2, 1, 3, 2, 2};
  for (int i = 0; i < 5; ++i) {
    launch_wconv(stream, P(90 + 4 * i),     w_hc[i], 64, 320, 9, 64);
    launch_wconv(stream, P(90 + 4 * i + 2), w_hf[i], hfOc[i], 64, 1, 16);
  }

  // ---- backbone branches (cbr) + map heads ----
  launch_conv(stream, xb, 256, 9, w_cc, 64, 64, P(3), P(4), P(5), P(6), P(2), 1, cf, 64, nullptr, 0, nullptr);
  launch_conv(stream, cf, 64, 9, w_chd, 16, 3, nullptr, nullptr, nullptr, nullptr, P(20), 0, nullptr, 0, nullptr, 0, out + OFF_CENTER);
  launch_conv(stream, xb, 256, 9, w_cnc, 64, 64, P(9), P(10), P(11), P(12), P(8), 1, cof, 64, nullptr, 0, nullptr);
  launch_conv(stream, cof, 64, 9, w_cnh, 16, 12, nullptr, nullptr, nullptr, nullptr, P(22), 0, nullptr, 0, nullptr, 0, out + OFF_CORNER);
  launch_conv(stream, xb, 256, 9, w_fgc, 64, 64, P(15), P(16), P(17), P(18), P(14), 1, ff, 64, nullptr, 0, nullptr);
  launch_conv(stream, ff, 64, 9, w_fgh, 16, 3, nullptr, nullptr, nullptr, nullptr, P(24), 0, nullptr, 0, nullptr, 0, out + OFF_FG);

  // ---- DCN fusion: offset_prep + 3x feature_adaption ----
  launch_conv(stream, xb, 256, 1, w_op, 64, 64, P(27), P(28), P(29), P(30), P(26), 1, om, 64, nullptr, 0, nullptr);
  const __bf16* dcnIn[3]  = {ff, cf, cof};
  __bf16*       dcnOut[3] = {fgf, ctrf, cnrf};
  for (int i = 0; i < 3; ++i) {
    launch_conv(stream, om, 64, 1, w_off[i], 32, 18, nullptr, nullptr, nullptr, nullptr,
                P(offW[i] + 1), 0, nullptr, 0, offb, 32, nullptr);
    dcn_wmma_k<<<NPIX / 128, 256, 0, stream>>>(dcnIn[i], offb, w_dcn[i], dcnOut[i]);
  }

  // ---- attention branches ----
  launch_conv(stream, xb, 256, 1, w_q, 16, 16, P(42), P(43), P(44), P(45), P(41), 1, qb, 16, nullptr, 0, nullptr);
  const __bf16* bFeat[3] = {ctrf, cnrf, fgf};
  float*        bA[3]    = {a_c, a_cn, a_f};
  __bf16*       bV[3]    = {v_c, v_cn, v_f};
  for (int i = 0; i < 3; ++i) {
    int w0 = kvW[i];
    launch_conv(stream, bFeat[i], 64, 1, w_k[i], 16, 16, P(w0 + 2), P(w0 + 3), P(w0 + 4), P(w0 + 5), P(w0 + 1), 1, kbuf, 16, nullptr, 0, nullptr);
    qk_dot_k<<<NPIX / 256, 256, 0, stream>>>(qb, kbuf, bA[i]);
    launch_conv(stream, bFeat[i], 64, 1, w_v[i], 64, 64, P(w0 + 8), P(w0 + 9), P(w0 + 10), P(w0 + 11), P(w0 + 7), 1, bV[i], 64, nullptr, 0, nullptr);
  }
  softmax_fuse_k<<<(unsigned)(((long)NPIX * 64 + 255) / 256), 256, 0, stream>>>(a_c, a_cn, a_f, v_c, v_cn, v_f, fused);

  // ---- fusion head ----
  launch_conv(stream, fused, 320, 9, w_fh0, 64, 64, P(84), P(85), P(86), P(87), P(83), 1, tbuf, 64, nullptr, 0, nullptr);
  launch_conv(stream, tbuf, 64, 9, w_fh1, 16, 3, nullptr, nullptr, nullptr, nullptr, P(89), 0, nullptr, 0, nullptr, 0, out + OFF_FUSION);

  // ---- 5 separated heads ----
  const long hOff[5] = {OFF_REG, OFF_HEIGHT, OFF_DIM, OFF_ROT, OFF_VEL};
  for (int i = 0; i < 5; ++i) {
    int w0 = 90 + 4 * i;
    launch_conv(stream, fused, 320, 9, w_hc[i], 64, 64, nullptr, nullptr, nullptr, nullptr, P(w0 + 1), 1, tbuf, 64, nullptr, 0, nullptr);
    launch_conv(stream, tbuf, 64, 1, w_hf[i], 16, hfOc[i], nullptr, nullptr, nullptr, nullptr, P(w0 + 3), 0, nullptr, 0, nullptr, 0, out + hOff[i]);
  }
}
